// DLRM_66099546685794
// MI455X (gfx1250) — matrix operand order repaired and checked by reading, not compiled, not run on hardware
//
#include <hip/hip_runtime.h>
#include <hip/hip_bf16.h>

typedef __attribute__((ext_vector_type(16))) __bf16 v16bf;
typedef __attribute__((ext_vector_type(8)))  float  v8f;
typedef unsigned short ushort_t;

// LDS row stride (elements) for 32-wide K tiles: 40 bf16 = 80 B.
// 80 B = 20 banks -> lane row-starts spaced 20 banks; over 16 lanes the
// starts are 4 banks apart (gcd(20,64)=4) => conflict-free ds_load_b128.
#define LPAD 40

// ---------- helpers ----------
__device__ __forceinline__ ushort_t f2bf(float f) {
    union { float f; unsigned u; } x; x.f = f;
    unsigned u = x.u + 0x7FFFu + ((x.u >> 16) & 1u);   // round-to-nearest-even
    return (ushort_t)(u >> 16);
}
__device__ __forceinline__ float bf2f(ushort_t h) {
    union { unsigned u; float f; } x; x.u = ((unsigned)h) << 16; return x.f;
}

// gfx1250 async global->LDS copy (ISA 15.18.3 op 98), tracked by ASYNCcnt.
// VDST = per-lane LDS byte offset, VADDR = 64-bit global address.
__device__ __forceinline__ void async_ld_b128(unsigned lds_off, const ushort_t* g) {
    asm volatile("global_load_async_to_lds_b128 %0, %1, off"
                 :: "v"(lds_off), "v"(g) : "memory");
}
#define WAIT_ASYNCCNT(n) asm volatile("s_wait_asynccnt " #n ::: "memory")

// ---------- prep kernels ----------
// num [B,16] f32 -> [B,32] bf16, zero-padded K 16..31
__global__ __launch_bounds__(256) void pad_num_bf16_kernel(
    const float* __restrict__ src, ushort_t* __restrict__ dst, int B)
{
    int i = blockIdx.x * 256 + threadIdx.x;
    if (i >= B * 32) return;
    int col = i & 31, row = i >> 5;
    dst[i] = (col < 16) ? f2bf(src[row * 16 + col]) : (ushort_t)0;
}

// W [K][N] f32 row-major -> WT [N][KP] bf16, zero-padded K..KP-1
__global__ __launch_bounds__(256) void transpose_pad_to_bf16_kernel(
    const float* __restrict__ W, ushort_t* __restrict__ WT, int K, int N, int KP)
{
    int i = blockIdx.x * 256 + threadIdx.x;
    if (i >= N * KP) return;
    int k = i % KP, n = i / KP;
    WT[i] = (k < K) ? f2bf(W[(size_t)k * N + n]) : (ushort_t)0;
}

// ---------- embedding gather: write bf16 into concat[:, 64 + 64*t] ----------
__global__ __launch_bounds__(256) void embed_gather_kernel(
    const int* __restrict__ cat,
    const float* __restrict__ t0, const float* __restrict__ t1,
    const float* __restrict__ t2, const float* __restrict__ t3,
    const float* __restrict__ t4, const float* __restrict__ t5,
    const float* __restrict__ t6, const float* __restrict__ t7,
    ushort_t* __restrict__ concat, int B)
{
    int gid = blockIdx.x * 256 + threadIdx.x;       // one thread = 4 floats
    if (gid >= B * 8 * 16) return;
    int chunk = gid & 15;            // 16 chunks of 4 = 64 elems
    int table = (gid >> 4) & 7;
    int row   = gid >> 7;
    const float* tbl;
    switch (table) {
        case 0: tbl = t0; break; case 1: tbl = t1; break;
        case 2: tbl = t2; break; case 3: tbl = t3; break;
        case 4: tbl = t4; break; case 5: tbl = t5; break;
        case 6: tbl = t6; break; default: tbl = t7; break;
    }
    int e = cat[row * 8 + table];
    float4 v = *(const float4*)(tbl + (size_t)e * 64 + chunk * 4);
    ushort_t* dst = concat + (size_t)row * 576 + 64 + table * 64 + chunk * 4;
    dst[0] = f2bf(v.x); dst[1] = f2bf(v.y); dst[2] = f2bf(v.z); dst[3] = f2bf(v.w);
}

// ---------- WMMA GEMM: C = act(A @ B + bias) ----------
// A: bf16 [M x K] row-major (lda = K, K % 32 == 0).
// BT: bf16 [N x K] row-major (B transposed, padded).
// C: bf16 (ldc). Block = 256 threads = 8 waves. Block tile M=128, N=64.
// Wave tile 32x32 = 2x2 WMMA fragments.
// Staging via GLOBAL_LOAD_ASYNC_TO_LDS_B128, software-pipelined one tile ahead.
// LDS rows padded to LPAD elements for conflict-free ds_load_b128.
__global__ __launch_bounds__(256) void wmma_gemm_bias_act_kernel(
    const ushort_t* __restrict__ A, const ushort_t* __restrict__ BT,
    const float* __restrict__ bias, ushort_t* __restrict__ C,
    int K, int ldc, int leaky)
{
    __shared__ __align__(16) ushort_t sA[2][128 * LPAD];  // 2 x 10 KB
    __shared__ __align__(16) ushort_t sB[2][64 * LPAD];   // 2 x 5 KB
    const int t    = threadIdx.x;
    const int wave = t >> 5;
    const int lane = t & 31;
    const int m0 = blockIdx.y * 128;
    const int n0 = blockIdx.x * 64;
    const int mo = (wave & 3) * 32;    // wave's M offset in block tile
    const int no = (wave >> 2) * 32;   // wave's N offset in block tile

    v8f acc[2][2] = {};

    // Staging maps (K % 32 == 0, pure b128 async copies):
    // A tile 128x32 bf16 = 512 b128 -> 2 per thread; B tile 64x32 = 256 b128 -> 1.
    const int ra0 = (t * 2) >> 2,     ca0 = ((t * 2) & 3) * 8;
    const int ra1 = (t * 2 + 1) >> 2, ca1 = ((t * 2 + 1) & 3) * 8;
    const int rb  = t >> 2,           cb  = (t & 3) * 8;

    const ushort_t* pa0 = A  + (size_t)(m0 + ra0) * K + ca0;
    const ushort_t* pa1 = A  + (size_t)(m0 + ra1) * K + ca1;
    const ushort_t* pb  = BT + (size_t)(n0 + rb)  * K + cb;

    // per-buffer LDS byte offsets for this thread's async destinations
    unsigned dA0[2], dA1[2], dB[2];
#pragma unroll
    for (int b = 0; b < 2; ++b) {
        dA0[b] = (unsigned)(size_t)&sA[b][ra0 * LPAD + ca0];
        dA1[b] = (unsigned)(size_t)&sA[b][ra1 * LPAD + ca1];
        dB[b]  = (unsigned)(size_t)&sB[b][rb  * LPAD + cb ];
    }

    const int nk = K >> 5;
    const int r  = lane & 15;
    const int h  = lane >> 4;

    // prologue: stage tile 0 into buffer 0
    async_ld_b128(dA0[0], pa0);
    async_ld_b128(dA1[0], pa1);
    async_ld_b128(dB[0],  pb);

    for (int ki = 0; ki < nk; ++ki) {
        const int buf = ki & 1;
        if (ki + 1 < nk) {
            const int k1 = (ki + 1) << 5;
            const int nb = buf ^ 1;
            async_ld_b128(dA0[nb], pa0 + k1);
            async_ld_b128(dA1[nb], pa1 + k1);
            async_ld_b128(dB[nb],  pb  + k1);
            if (ki + 2 < nk) {  // warm L2 two tiles ahead (global_prefetch_b8)
                __builtin_prefetch(pa0 + k1 + 32, 0, 1);
                __builtin_prefetch(pb  + k1 + 32, 0, 1);
            }
            WAIT_ASYNCCNT(0x3);   // in-order: tile ki's 3 asyncs have landed
        } else {
            WAIT_ASYNCCNT(0x0);
        }
        __syncthreads();

        // --- fragments per ISA 7.12.2 layouts ---
        union frag { v16bf v; uint4 q[2]; };
        frag a[2], b[2];
#pragma unroll
        for (int i = 0; i < 2; ++i) {
            // A 16x32: lane<16 -> M=lane, K {0..7,16..23}; lane>=16 -> K {8..15,24..31}
            const ushort_t* pA = &sA[buf][(mo + i * 16 + r) * LPAD + h * 8];
            a[i].q[0] = *(const uint4*)pA;
            a[i].q[1] = *(const uint4*)(pA + 16);
            // B 32x16 (BT layout): lane<16 -> N=lane, K {0..7,16..23}; lane>=16 -> K {8..15,24..31}
            const ushort_t* pB = &sB[buf][(no + i * 16 + r) * LPAD + h * 8];
            b[i].q[0] = *(const uint4*)pB;
            b[i].q[1] = *(const uint4*)(pB + 16);
        }
#pragma unroll
        for (int i = 0; i < 2; ++i)
#pragma unroll
            for (int j = 0; j < 2; ++j)
                acc[i][j] = __builtin_amdgcn_wmma_f32_16x16x32_bf16(
                    false, a[i].v, false, b[j].v, (short)0, acc[i][j], false, false);

        __syncthreads();   // all waves done reading buf before tile ki+2 overwrites it
    }

    // --- epilogue: D layout lane<16 -> M=r8, lane>=16 -> M=r8+8; N=lane&15 ---
    const int c = lane & 15, hh = lane >> 4;
#pragma unroll
    for (int j = 0; j < 2; ++j) {
        int gcol = n0 + no + j * 16 + c;
        float bb = bias[gcol];
#pragma unroll
        for (int i = 0; i < 2; ++i) {
#pragma unroll
            for (int r8 = 0; r8 < 8; ++r8) {
                int gm = m0 + mo + i * 16 + r8 + 8 * hh;
                float v = acc[i][j][r8] + bb;
                if (leaky) v = (v > 0.0f) ? v : 0.01f * v;
                C[(size_t)gm * ldc + gcol] = f2bf(v);
            }
        }
    }
}

// ---------- final layer: [B,256] @ [256,1] + b, sigmoid ----------
__global__ __launch_bounds__(256) void final_layer_kernel(
    const ushort_t* __restrict__ z2, const float* __restrict__ w,
    const float* __restrict__ b, float* __restrict__ out, int B)
{
    int wave = threadIdx.x >> 5;
    int lane = threadIdx.x & 31;
    int row  = blockIdx.x * 8 + wave;
    if (row >= B) return;
    const ushort_t* zr = z2 + (size_t)row * 256;
    float acc = 0.0f;
#pragma unroll
    for (int j = 0; j < 8; ++j) {
        int k = lane + j * 32;
        acc += bf2f(zr[k]) * w[k];
    }
#pragma unroll
    for (int s = 16; s > 0; s >>= 1)
        acc += __shfl_down(acc, s, 32);
    if (lane == 0) {
        float v = acc + b[0];
        out[row] = 1.0f / (1.0f + __expf(-v));
    }
}

// ---------- launcher ----------
extern "C" void kernel_launch(void* const* d_in, const int* in_sizes, int n_in,
                              void* d_out, int out_size, void* d_ws, size_t ws_size,
                              hipStream_t stream) {
    (void)in_sizes; (void)n_in; (void)out_size; (void)ws_size;
    const int B = 16384;

    const float* num   = (const float*)d_in[0];   // [B,16]
    const int*   cat   = (const int*)  d_in[1];   // [B,8]
    const float* botW0 = (const float*)d_in[2];   // [16,128]
    const float* botW1 = (const float*)d_in[3];   // [128,64]
    const float* botB0 = (const float*)d_in[4];   // [128]
    const float* botB1 = (const float*)d_in[5];   // [64]
    const float* emb0  = (const float*)d_in[6];
    const float* emb1  = (const float*)d_in[7];
    const float* emb2  = (const float*)d_in[8];
    const float* emb3  = (const float*)d_in[9];
    const float* emb4  = (const float*)d_in[10];
    const float* emb5  = (const float*)d_in[11];
    const float* emb6  = (const float*)d_in[12];
    const float* emb7  = (const float*)d_in[13];
    const float* topW0 = (const float*)d_in[14];  // [576,512]
    const float* topW1 = (const float*)d_in[15];  // [512,256]
    const float* topW2 = (const float*)d_in[16];  // [256,1] -> plain vector
    const float* topB0 = (const float*)d_in[17];  // [512]
    const float* topB1 = (const float*)d_in[18];  // [256]
    const float* topB2 = (const float*)d_in[19];  // [1]
    float* out = (float*)d_out;

    // workspace carve (256B aligned)
    char* ws = (char*)d_ws; size_t off = 0;
    auto carve = [&](size_t bytes) -> void* {
        void* p = ws + off; off += (bytes + 255) & ~(size_t)255; return p;
    };
    ushort_t* numb = (ushort_t*)carve((size_t)B * 32 * 2);      // padded K=32
    ushort_t* w0T  = (ushort_t*)carve((size_t)128 * 32 * 2);    // padded K=32
    ushort_t* w1T  = (ushort_t*)carve((size_t)64 * 128 * 2);
    ushort_t* tw0T = (ushort_t*)carve((size_t)512 * 576 * 2);
    ushort_t* tw1T = (ushort_t*)carve((size_t)256 * 512 * 2);
    ushort_t* x1   = (ushort_t*)carve((size_t)B * 128 * 2);
    ushort_t* cc   = (ushort_t*)carve((size_t)B * 576 * 2);
    ushort_t* z1   = (ushort_t*)carve((size_t)B * 512 * 2);
    ushort_t* z2   = (ushort_t*)carve((size_t)B * 256 * 2);

    // 1) precision conversion / weight transpose (K padded to mult. of 32)
    pad_num_bf16_kernel<<<(B * 32 + 255) / 256, 256, 0, stream>>>(num, numb, B);
    transpose_pad_to_bf16_kernel<<<(128 * 32 + 255) / 256, 256, 0, stream>>>(botW0, w0T, 16, 128, 32);
    transpose_pad_to_bf16_kernel<<<(64 * 128 + 255) / 256, 256, 0, stream>>>(botW1, w1T, 128, 64, 128);
    transpose_pad_to_bf16_kernel<<<(512 * 576 + 255) / 256, 256, 0, stream>>>(topW0, tw0T, 576, 512, 576);
    transpose_pad_to_bf16_kernel<<<(256 * 512 + 255) / 256, 256, 0, stream>>>(topW1, tw1T, 512, 256, 512);

    // 2) embedding gather -> concat cols 64..575
    embed_gather_kernel<<<(B * 8 * 16 + 255) / 256, 256, 0, stream>>>(
        cat, emb0, emb1, emb2, emb3, emb4, emb5, emb6, emb7, cc, B);

    // 3) bottom MLP (leaky both layers); layer 2 writes concat cols 0..63
    wmma_gemm_bias_act_kernel<<<dim3(128 / 64, B / 128), 256, 0, stream>>>(
        numb, w0T, botB0, x1, /*K=*/32, /*ldc=*/128, 1);
    wmma_gemm_bias_act_kernel<<<dim3(64 / 64, B / 128), 256, 0, stream>>>(
        x1, w1T, botB1, cc, /*K=*/128, /*ldc=*/576, 1);

    // 4) top MLP layers 1,2 (leaky)
    wmma_gemm_bias_act_kernel<<<dim3(512 / 64, B / 128), 256, 0, stream>>>(
        cc, tw0T, topB0, z1, /*K=*/576, /*ldc=*/512, 1);
    wmma_gemm_bias_act_kernel<<<dim3(256 / 64, B / 128), 256, 0, stream>>>(
        z1, tw1T, topB1, z2, /*K=*/512, /*ldc=*/256, 1);

    // 5) final 256->1 + sigmoid
    final_layer_kernel<<<B / 8, 256, 0, stream>>>(z2, topW2, topB2, out, B);
}
